// MaskedHeteroGAT_35124242546830
// MI455X (gfx1250) — matrix operand order repaired and checked by reading, not compiled, not run on hardware
//
#include <hip/hip_runtime.h>
#include <cstdint>
#include <cstddef>

typedef __attribute__((ext_vector_type(16))) _Float16 v16h;
typedef __attribute__((ext_vector_type(8)))  float    v8f;
typedef __attribute__((ext_vector_type(4)))  _Float16 h4;

#define NP   20000   // packages
#define NTG  30000   // targets per type
#define NE   100000  // edges per type
#define NTYP 6
#define INC  400
#define D1   256
#define KC   128
#define HH   4

// ---------------- wave helpers (wave32) ----------------
__device__ inline float wred_sum(float v){
#pragma unroll
  for (int m = 16; m > 0; m >>= 1) v += __shfl_xor(v, m, 32);
  return v;
}
__device__ inline float wred_max(float v){
#pragma unroll
  for (int m = 16; m > 0; m >>= 1) v = fmaxf(v, __shfl_xor(v, m, 32));
  return v;
}
__device__ inline void atomicMaxF(float* a, float v){
  if (v >= 0.f) atomicMax((int*)a, __float_as_int(v));
  else          atomicMin((unsigned int*)a, __float_as_uint(v));
}
__device__ inline float lrelu(float x){ return x > 0.f ? x : 0.2f * x; }

// ---------------- shared WMMA compute: 128x64 tile, 4 waves ----------------
// As: 128 rows x 32 k (stride 40 halves); Bs: 64 cols x 32 k (stride 40).
// Wave w covers m in [w*32, w*32+32): two A fragments, B fragment reused.
__device__ __forceinline__ void mma_128x64(const _Float16* As, const _Float16* Bs,
                                           int wave, int lane, v8f acc[2][4])
{
  const int half = lane >> 4, l15 = lane & 15;
  const int kb = half * 16;
  const int mr0 = wave * 32 + l15;
  const int mr1 = mr0 + 16;
  v16h a0, a1;
#pragma unroll
  for (int v = 0; v < 8; ++v) {
    int bk = ((v >> 2) << 4) + (half << 3) + ((v & 3) << 1);
    a0[2 * v]     = As[mr0 * 40 + bk];
    a0[2 * v + 1] = As[mr0 * 40 + bk + 1];
    a1[2 * v]     = As[mr1 * 40 + bk];
    a1[2 * v + 1] = As[mr1 * 40 + bk + 1];
  }
#pragma unroll
  for (int j = 0; j < 4; ++j) {
    int nc = j * 16 + l15;
    v16h b;
#pragma unroll
    for (int v = 0; v < 8; ++v) {
      int bk = ((v >> 2) << 4) + (half << 3) + ((v & 3) << 1);
      b[2 * v]     = Bs[nc * 40 + bk];
      b[2 * v + 1] = Bs[nc * 40 + bk + 1];
    }
    acc[0][j] = __builtin_amdgcn_wmma_f32_16x16x32_f16(false, a0, false, b,
                                                       (short)0, acc[0][j], false, false);
    acc[1][j] = __builtin_amdgcn_wmma_f32_16x16x32_f16(false, a1, false, b,
                                                       (short)0, acc[1][j], false, false);
  }
}

// =======================================================================
// WMMA GEMM: C[M,N] = A[M,K] @ B[K,N]  (f32 in/out, f16 compute)
// block = 128 threads, tile 128x64, K-step 32. Main loop is branch-free
// (A rows clamped; garbage rows masked at store); ragged K tail peeled.
// N must be a multiple of 64 (always 128/256 here).
// =======================================================================
__global__ void gemm_nn_k(const float* __restrict__ A, const float* __restrict__ B,
                          float* __restrict__ C, int M, int N, int K, int do_relu)
{
  __shared__ _Float16 As[128 * 40];
  __shared__ _Float16 Bs[64 * 40];
  const int tid = (int)threadIdx.x;
  const int bM = blockIdx.y * 128, bN = blockIdx.x * 64;
  const int lane = tid & 31, wave = tid >> 5;
  const int a_ks = (tid & 7) * 4;       // A: 8 lanes x float4 per row
  const int a_r0 = tid >> 3;            // 16 rows per pass, 8 passes
  const int b_ns = (tid & 15) * 4;      // B: 16 lanes x float4 over n
  const int b_c0 = tid >> 4;            // 8 k per pass, 4 passes

  v8f acc[2][4] = {};
  const int Kfull = K & ~31;

  for (int k0 = 0; k0 < Kfull; k0 += 32) {   // ---- branch-free main loop ----
#pragma unroll
    for (int i = 0; i < 8; ++i) {
      int r = a_r0 + i * 16;
      int gr = bM + r; if (gr >= M) gr = M - 1;            // clamp (masked at store)
      float4 val = *(const float4*)(A + (size_t)gr * K + k0 + a_ks);
      h4 hv; hv.x = (_Float16)val.x; hv.y = (_Float16)val.y;
             hv.z = (_Float16)val.z; hv.w = (_Float16)val.w;
      *(h4*)&As[r * 40 + a_ks] = hv;
    }
#pragma unroll
    for (int i = 0; i < 4; ++i) {
      int c = b_c0 + i * 8;
      float4 val = *(const float4*)(B + (size_t)(k0 + c) * N + bN + b_ns);
      Bs[(b_ns + 0) * 40 + c] = (_Float16)val.x;
      Bs[(b_ns + 1) * 40 + c] = (_Float16)val.y;
      Bs[(b_ns + 2) * 40 + c] = (_Float16)val.z;
      Bs[(b_ns + 3) * 40 + c] = (_Float16)val.w;
    }
    __syncthreads();
    mma_128x64(As, Bs, wave, lane, acc);
    __syncthreads();
  }

  if (Kfull < K) {                            // ---- ragged K tail, once ----
#pragma unroll
    for (int i = 0; i < 8; ++i) {
      int r = a_r0 + i * 16;
      int gr = bM + r; if (gr >= M) gr = M - 1;
      const float* p = A + (size_t)gr * K;
      int gc = Kfull + a_ks;
      h4 hv;
#pragma unroll
      for (int q = 0; q < 4; ++q) {
        int g = gc + q;
        int gcl = g < K ? g : K - 1;          // clamped load, select to zero
        float x = p[gcl];
        hv[q] = (_Float16)(g < K ? x : 0.f);
      }
      *(h4*)&As[r * 40 + a_ks] = hv;
    }
#pragma unroll
    for (int i = 0; i < 4; ++i) {
      int c = b_c0 + i * 8;
      int gk = Kfull + c;
      int gkc = gk < K ? gk : K - 1;
      float4 val = *(const float4*)(B + (size_t)gkc * N + bN + b_ns);
      float msk = gk < K ? 1.f : 0.f;
      Bs[(b_ns + 0) * 40 + c] = (_Float16)(val.x * msk);
      Bs[(b_ns + 1) * 40 + c] = (_Float16)(val.y * msk);
      Bs[(b_ns + 2) * 40 + c] = (_Float16)(val.z * msk);
      Bs[(b_ns + 3) * 40 + c] = (_Float16)(val.w * msk);
    }
    __syncthreads();
    mma_128x64(As, Bs, wave, lane, acc);
  }

  // C/D layout: VGPR v -> row = half*8 + v, col = l15
  const int half = lane >> 4, l15 = lane & 15;
  if (bM + 128 <= M) {
#pragma unroll
    for (int u = 0; u < 2; ++u)
#pragma unroll
      for (int j = 0; j < 4; ++j)
#pragma unroll
        for (int v = 0; v < 8; ++v) {
          int row = bM + wave * 32 + u * 16 + half * 8 + v;
          int col = bN + j * 16 + l15;
          float x = acc[u][j][v];
          if (do_relu) x = fmaxf(x, 0.f);
          C[(size_t)row * N + col] = x;
        }
  } else {
#pragma unroll
    for (int u = 0; u < 2; ++u)
#pragma unroll
      for (int j = 0; j < 4; ++j)
#pragma unroll
        for (int v = 0; v < 8; ++v) {
          int row = bM + wave * 32 + u * 16 + half * 8 + v;
          int col = bN + j * 16 + l15;
          if (row < M) {
            float x = acc[u][j][v];
            if (do_relu) x = fmaxf(x, 0.f);
            C[(size_t)row * N + col] = x;
          }
        }
  }
}

// =======================================================================
// WMMA GEMM, A transposed storage + optional row gather on both operands:
// C[M,N] = sum_k A[ia[k], m] * B[ib[k], n]; A stored [K',M], B stored [K',N]
// M must be a multiple of 128 (always 128 here), N a multiple of 64.
// =======================================================================
__global__ void gemm_tn_k(const float* __restrict__ A, const float* __restrict__ B,
                          float* __restrict__ C, int M, int N, int K,
                          const int* __restrict__ ia, const int* __restrict__ ib)
{
  __shared__ _Float16 As[128 * 40];
  __shared__ _Float16 Bs[64 * 40];
  const int tid = (int)threadIdx.x;
  const int bM = blockIdx.y * 128, bN = blockIdx.x * 64;
  const int lane = tid & 31, wave = tid >> 5;
  const int seg = (tid & 15) * 4;       // 4 consecutive m (or n)
  const int c0  = tid >> 4;             // 0..7, k within tile

  v8f acc[2][4] = {};
  const int Kfull = K & ~31;

  for (int k0 = 0; k0 < Kfull; k0 += 32) {   // ---- branch-free main loop ----
#pragma unroll
    for (int i = 0; i < 8; ++i) {            // A: 8 passes = 4 k-groups x 2 m-halves
      int c = c0 + (i >> 1) * 8;
      int moff = (i & 1) * 64;
      int kk = k0 + c;
      int ra = ia ? ia[kk] : kk;
      float4 val = *(const float4*)(A + (size_t)ra * M + bM + moff + seg);
      As[(moff + seg + 0) * 40 + c] = (_Float16)val.x;
      As[(moff + seg + 1) * 40 + c] = (_Float16)val.y;
      As[(moff + seg + 2) * 40 + c] = (_Float16)val.z;
      As[(moff + seg + 3) * 40 + c] = (_Float16)val.w;
    }
#pragma unroll
    for (int i = 0; i < 4; ++i) {
      int c = c0 + i * 8;
      int kk = k0 + c;
      int rb = ib ? ib[kk] : kk;
      float4 val = *(const float4*)(B + (size_t)rb * N + bN + seg);
      Bs[(seg + 0) * 40 + c] = (_Float16)val.x;
      Bs[(seg + 1) * 40 + c] = (_Float16)val.y;
      Bs[(seg + 2) * 40 + c] = (_Float16)val.z;
      Bs[(seg + 3) * 40 + c] = (_Float16)val.w;
    }
    __syncthreads();
    mma_128x64(As, Bs, wave, lane, acc);
    __syncthreads();
  }

  if (Kfull < K) {                            // ---- ragged K tail, once ----
#pragma unroll
    for (int i = 0; i < 8; ++i) {
      int c = c0 + (i >> 1) * 8;
      int moff = (i & 1) * 64;
      int kk = Kfull + c;
      int kkc = kk < K ? kk : K - 1;
      int ra = ia ? ia[kkc] : kkc;
      float4 val = *(const float4*)(A + (size_t)ra * M + bM + moff + seg);
      float msk = kk < K ? 1.f : 0.f;
      As[(moff + seg + 0) * 40 + c] = (_Float16)(val.x * msk);
      As[(moff + seg + 1) * 40 + c] = (_Float16)(val.y * msk);
      As[(moff + seg + 2) * 40 + c] = (_Float16)(val.z * msk);
      As[(moff + seg + 3) * 40 + c] = (_Float16)(val.w * msk);
    }
#pragma unroll
    for (int i = 0; i < 4; ++i) {
      int c = c0 + i * 8;
      int kk = Kfull + c;
      int kkc = kk < K ? kk : K - 1;
      int rb = ib ? ib[kkc] : kkc;
      float4 val = *(const float4*)(B + (size_t)rb * N + bN + seg);
      float msk = kk < K ? 1.f : 0.f;
      Bs[(seg + 0) * 40 + c] = (_Float16)(val.x * msk);
      Bs[(seg + 1) * 40 + c] = (_Float16)(val.y * msk);
      Bs[(seg + 2) * 40 + c] = (_Float16)(val.z * msk);
      Bs[(seg + 3) * 40 + c] = (_Float16)(val.w * msk);
    }
    __syncthreads();
    mma_128x64(As, Bs, wave, lane, acc);
  }

  const int half = lane >> 4, l15 = lane & 15;
#pragma unroll
  for (int u = 0; u < 2; ++u)
#pragma unroll
    for (int j = 0; j < 4; ++j)
#pragma unroll
      for (int v = 0; v < 8; ++v) {
        int row = bM + wave * 32 + u * 16 + half * 8 + v;
        int col = bN + j * 16 + l15;
        C[(size_t)row * N + col] = acc[u][j][v];
      }
}

// ---------------- elementwise / utility kernels ----------------
__global__ void fill_k(float* p, float v, long n){
  long i = (long)blockIdx.x * blockDim.x + threadIdx.x;
  if (i < n) p[i] = v;
}
__global__ void relu_k(float* p, long n){
  long i = (long)blockIdx.x * blockDim.x + threadIdx.x;
  if (i < n) p[i] = fmaxf(p[i], 0.f);
}
__global__ void fix_emax_k(float* p, int n){
  int i = blockIdx.x * blockDim.x + threadIdx.x;
  if (i < n) { float v = p[i]; if (!(v > -1e38f)) p[i] = 0.f; }  // -inf / NaN -> 0
}
__global__ void write_scalar_k(const float* src, float* dst){
  if (threadIdx.x == 0 && blockIdx.x == 0) dst[0] = src[0];
}

// row softmax over 128 cols, in place; accumulates mean entropy into *loss.
__global__ void softmax128_ent_k(float* __restrict__ S, int n, float inv_n, float* loss)
{
  int lane = threadIdx.x & 31, wave = threadIdx.x >> 5;
  int row = blockIdx.x * 8 + wave;
  if (row >= n) return;
  float* p = S + (size_t)row * 128;
  float v[4];
#pragma unroll
  for (int k = 0; k < 4; ++k) v[k] = p[lane + 32 * k];
  float m = wred_max(fmaxf(fmaxf(v[0], v[1]), fmaxf(v[2], v[3])));
  float s = 0.f;
#pragma unroll
  for (int k = 0; k < 4; ++k) { v[k] = __expf(v[k] - m); s += v[k]; }
  s = wred_sum(s);
  float inv = 1.f / s, ent = 0.f;
#pragma unroll
  for (int k = 0; k < 4; ++k) {
    float pr = v[k] * inv;
    p[lane + 32 * k] = pr;
    ent -= pr * __logf(pr + 1e-15f);
  }
  ent = wred_sum(ent);
  if (lane == 0) atomicAdd(loss, ent * inv_n);
}

// per-edge GATv2 scores: wave per edge; lane owns 8 contiguous channels
// (one head per 8-lane group).
__global__ void edge_score_k(const float* __restrict__ l, const float* __restrict__ r,
                             const int* __restrict__ src, const int* __restrict__ tgt,
                             const float* __restrict__ att, float* __restrict__ e,
                             float* __restrict__ emax, int E)
{
  int lane = threadIdx.x & 31, wave = threadIdx.x >> 5;
  int ed = blockIdx.x * 8 + wave;
  if (ed >= E) return;
  int s = src[ed], t = tgt[ed];
  int h = lane >> 3, base = lane * 8;
  const float* lp = l + (size_t)s * 256 + base;
  const float* rp = r + (size_t)t * 256 + base;
  const float* ap = att + base;
  float acc = 0.f;
#pragma unroll
  for (int q = 0; q < 8; ++q) acc += ap[q] * lrelu(lp[q] + rp[q]);
  acc += __shfl_xor(acc, 1, 32);
  acc += __shfl_xor(acc, 2, 32);
  acc += __shfl_xor(acc, 4, 32);
  if ((lane & 7) == 0) {
    e[(size_t)ed * 4 + h] = acc;
    atomicMaxF(&emax[t * 4 + h], acc);
  }
}

__global__ void edge_exp_k(const float* __restrict__ e, const float* __restrict__ emax,
                           const int* __restrict__ tgt, float* __restrict__ den, int E)
{
  long g = (long)blockIdx.x * blockDim.x + threadIdx.x;
  if (g >= (long)E * 4) return;
  int ed = (int)(g >> 2), h = (int)(g & 3);
  int t = tgt[ed];
  atomicAdd(&den[t * 4 + h], __expf(e[g] - emax[t * 4 + h]));
}

__global__ void edge_scatter_k(const float* __restrict__ l, const float* __restrict__ e,
                               const float* __restrict__ emax, const float* __restrict__ den,
                               const int* __restrict__ src, const int* __restrict__ tgt,
                               float* __restrict__ agg, int E)
{
  int lane = threadIdx.x & 31, wave = threadIdx.x >> 5;
  int ed = blockIdx.x * 8 + wave;
  if (ed >= E) return;
  int s = src[ed], t = tgt[ed];
  int h = lane >> 3, base = lane * 8;
  float al = __expf(e[(size_t)ed * 4 + h] - emax[t * 4 + h]) / (den[t * 4 + h] + 1e-16f);
  const float* lp = l + (size_t)s * 256 + base;
  float* op = agg + (size_t)t * 256 + base;
#pragma unroll
  for (int q = 0; q < 8; ++q) atomicAdd(&op[q], lp[q] * al);
}

// link loss: sqrt(relu(E - 2*trace(A_pool) + sum(M_pkg*Mt)) + 1e-12) * scale
__global__ void link_loss_k(const float* __restrict__ Apool, const float* __restrict__ Mp,
                            const float* __restrict__ Mt, float n_edges, float scale,
                            float* loss)
{
  __shared__ float red[256];
  int tid = threadIdx.x;
  float acc = 0.f;
  for (int i = tid; i < 128 * 128; i += 256) acc += Mp[i] * Mt[i];
  float tr = 0.f;
  for (int i = tid; i < 128; i += 256) tr += Apool[i * 128 + i];
  red[tid] = acc; __syncthreads();
  for (int s2 = 128; s2 > 0; s2 >>= 1) { if (tid < s2) red[tid] += red[tid + s2]; __syncthreads(); }
  float mm = red[0]; __syncthreads();
  red[tid] = tr; __syncthreads();
  for (int s2 = 128; s2 > 0; s2 >>= 1) { if (tid < s2) red[tid] += red[tid + s2]; __syncthreads(); }
  if (tid == 0) {
    float norm2 = n_edges - 2.f * red[0] + mm;
    atomicAdd(loss, sqrtf(fmaxf(norm2, 0.f) + 1e-12f) * scale);
  }
}

// dense pooled GATv2 attention, masked softmax over source axis.
// block per target cluster j (128 blocks), 4 waves = 4 heads.
__global__ void attn2_k(const float* __restrict__ l2, const float* __restrict__ r2,
                        const float* __restrict__ Apool, const float* __restrict__ att,
                        float* __restrict__ outAttn)
{
  __shared__ float es[4 * 128];
  int j = blockIdx.x;
  int lane = threadIdx.x & 31, h = threadIdx.x >> 5;
  const float* ap = att + h * 64;
  const float* rp = r2 + (size_t)j * 256 + h * 64;
  float r0 = rp[lane], r1 = rp[lane + 32];
  float a0 = ap[lane], a1 = ap[lane + 32];
  for (int i = 0; i < 128; ++i) {
    const float* lp = l2 + (size_t)i * 256 + h * 64;
    float acc = a0 * lrelu(lp[lane] + r0) + a1 * lrelu(lp[lane + 32] + r1);
    acc = wred_sum(acc);
    if (lane == 0) es[h * 128 + i] = (Apool[i * 128 + j] > 0.f) ? acc : -1e9f;
  }
  __syncthreads();
  float v[4];
#pragma unroll
  for (int k = 0; k < 4; ++k) v[k] = es[h * 128 + lane + 32 * k];
  float m = wred_max(fmaxf(fmaxf(v[0], v[1]), fmaxf(v[2], v[3])));
  float s = 0.f;
#pragma unroll
  for (int k = 0; k < 4; ++k) { v[k] = __expf(v[k] - m); s += v[k]; }
  s = wred_sum(s);
  float inv = 1.f / s;
#pragma unroll
  for (int k = 0; k < 4; ++k) {
    int i = lane + 32 * k;
    outAttn[((size_t)i * 128 + j) * 4 + h] = v[k] * inv;   // [i, j, h]
  }
}

// classifier: sigmoid(pooled @ Wcls + b); one wave per row (896 rows).
__global__ void classifier_k(const float* __restrict__ pooled, const float* __restrict__ Wc,
                             const float* __restrict__ bc, float* __restrict__ out)
{
  int lane = threadIdx.x & 31, wave = threadIdx.x >> 5;
  int row = blockIdx.x * 8 + wave;
  if (row >= 7 * 128) return;
  float acc = 0.f;
#pragma unroll
  for (int k = 0; k < 8; ++k) {
    int c = lane + 32 * k;
    acc += pooled[(size_t)row * 256 + c] * Wc[c];
  }
  acc = wred_sum(acc);
  if (lane == 0) out[row] = 1.f / (1.f + __expf(-(acc + bc[0])));
}

// ---------------- host-side launch helpers ----------------
static void gemm_nn(hipStream_t s, const float* A, const float* B, float* C,
                    int M, int N, int K, int relu){
  dim3 g((unsigned)((N + 63) / 64), (unsigned)((M + 127) / 128));
  gemm_nn_k<<<g, 128, 0, s>>>(A, B, C, M, N, K, relu);
}
static void gemm_tn(hipStream_t s, const float* A, const float* B, float* C,
                    int M, int N, int K, const int* ia, const int* ib){
  dim3 g((unsigned)((N + 63) / 64), (unsigned)((M + 127) / 128));
  gemm_tn_k<<<g, 128, 0, s>>>(A, B, C, M, N, K, ia, ib);
}

extern "C" void kernel_launch(void* const* d_in, const int* in_sizes, int n_in,
                              void* d_out, int out_size, void* d_ws, size_t ws_size,
                              hipStream_t stream)
{
  (void)in_sizes; (void)n_in; (void)out_size; (void)ws_size;
  const float* x_pkg   = (const float*)d_in[0];
  const float* x_tgt   = (const float*)d_in[1];
  const int*   src_idx = (const int*)d_in[2];
  const int*   tgt_idx = (const int*)d_in[3];
  const float* W1s     = (const float*)d_in[4];
  const float* W1t     = (const float*)d_in[5];
  const float* a1      = (const float*)d_in[6];
  const float* Wpkg    = (const float*)d_in[7];
  const float* Wassign = (const float*)d_in[8];
  const float* W2s     = (const float*)d_in[9];
  const float* W2t     = (const float*)d_in[10];
  const float* a2      = (const float*)d_in[11];
  const float* Wcls    = (const float*)d_in[12];
  const float* bcls    = (const float*)d_in[13];
  float* out = (float*)d_out;

  // workspace carve (~132 MB of f32); per-type buffers reused across types.
  float* w = (float*)d_ws;
  auto take = [&](size_t n){ float* p = w; w += n; return p; };
  float* h_pkg  = take((size_t)NP * D1);
  float* S_pkg  = take((size_t)NP * KC);
  float* pooled = take((size_t)7 * KC * D1);   // p_pkg then p_tgt[0..5]
  float* M_pkg  = take((size_t)KC * KC);
  float* loss   = take(1);
  float* l_pkg  = take((size_t)NP * D1);
  float* r_tgt  = take((size_t)NTG * D1);
  float* agg    = take((size_t)NTG * D1);      // aggregation -> h_tgt
  float* S_tgt  = take((size_t)NTG * KC);
  float* ebuf   = take((size_t)NE * HH);
  float* emax   = take((size_t)NTG * HH);
  float* den    = take((size_t)NTG * HH);
  float* A_pool = take((size_t)KC * KC);
  float* Mt     = take((size_t)KC * KC);
  float* l2     = take((size_t)KC * D1);
  float* r2     = take((size_t)KC * D1);

  fill_k<<<1, 1, 0, stream>>>(loss, 0.f, 1);

  // h_pkg = relu(x_pkg @ Wpkg)
  gemm_nn(stream, x_pkg, Wpkg, h_pkg, NP, D1, INC, 1);
  // S_pkg = softmax(h_pkg @ Wassign[0]) (+ entropy into loss)
  gemm_nn(stream, h_pkg, Wassign, S_pkg, NP, KC, D1, 0);
  softmax128_ent_k<<<NP / 8, 256, 0, stream>>>(S_pkg, NP, 1.f / (float)NP, loss);
  // p_pkg = S_pkg^T @ h_pkg ; M_pkg = S_pkg^T @ S_pkg
  gemm_tn(stream, S_pkg, h_pkg, pooled, KC, D1, NP, nullptr, nullptr);
  gemm_tn(stream, S_pkg, S_pkg, M_pkg, KC, KC, NP, nullptr, nullptr);

  for (int t = 0; t < NTYP; ++t) {
    const float* xt = x_tgt + (size_t)t * NTG * INC;
    const int*   st = src_idx + (size_t)t * NE;
    const int*   tt = tgt_idx + (size_t)t * NE;

    // conv1 projections
    gemm_nn(stream, x_pkg, W1s + (size_t)t * INC * D1, l_pkg, NP, D1, INC, 0);
    gemm_nn(stream, xt,    W1t + (size_t)t * INC * D1, r_tgt, NTG, D1, INC, 0);

    fill_k<<<(NTG * HH + 255) / 256, 256, 0, stream>>>(emax, -__builtin_inff(), (long)NTG * HH);
    fill_k<<<(NTG * HH + 255) / 256, 256, 0, stream>>>(den, 0.f, (long)NTG * HH);
    fill_k<<<(int)(((long)NTG * D1 + 255) / 256), 256, 0, stream>>>(agg, 0.f, (long)NTG * D1);

    // per-edge scores, segment softmax, weighted scatter
    edge_score_k<<<NE / 8, 256, 0, stream>>>(l_pkg, r_tgt, st, tt,
                                             a1 + (size_t)t * HH * 64, ebuf, emax, NE);
    fix_emax_k<<<(NTG * HH + 255) / 256, 256, 0, stream>>>(emax, NTG * HH);
    edge_exp_k<<<(NE * HH + 255) / 256, 256, 0, stream>>>(ebuf, emax, tt, den, NE);
    edge_scatter_k<<<NE / 8, 256, 0, stream>>>(l_pkg, ebuf, emax, den, st, tt, agg, NE);
    relu_k<<<(int)(((long)NTG * D1 + 255) / 256), 256, 0, stream>>>(agg, (long)NTG * D1); // h_tgt

    // diffpool for this type
    gemm_nn(stream, agg, Wassign + (size_t)(t + 1) * D1 * KC, S_tgt, NTG, KC, D1, 0);
    softmax128_ent_k<<<NTG / 8, 256, 0, stream>>>(S_tgt, NTG, 1.f / (float)NTG, loss);
    gemm_tn(stream, S_tgt, agg, pooled + (size_t)(t + 1) * KC * D1, KC, D1, NTG, nullptr, nullptr);
    gemm_tn(stream, S_pkg, S_tgt, A_pool, KC, KC, NE, st, tt);     // gathered pooled adjacency
    gemm_tn(stream, S_tgt, S_tgt, Mt, KC, KC, NTG, nullptr, nullptr);
    link_loss_k<<<1, 256, 0, stream>>>(A_pool, M_pkg, Mt, (float)NE,
                                       1.f / ((float)NP * (float)NTG), loss);

    // conv2 dense attention over pooled graph
    gemm_nn(stream, pooled, W2s + (size_t)t * D1 * D1, l2, KC, D1, D1, 0);
    gemm_nn(stream, pooled + (size_t)(t + 1) * KC * D1, W2t + (size_t)t * D1 * D1, r2, KC, D1, D1, 0);
    attn2_k<<<KC, 128, 0, stream>>>(l2, r2, A_pool, a2 + (size_t)t * HH * 64,
                                    out + 897 + (size_t)t * KC * KC * HH);
  }

  // probs over all pooled nodes; loss scalar at out[896]
  classifier_k<<<(7 * 128) / 8, 256, 0, stream>>>(pooled, Wcls, bcls, out);
  write_scalar_k<<<1, 1, 0, stream>>>(loss, out + 896);
}
